// GATBlock_6768868459348
// MI455X (gfx1250) — matrix operand order repaired and checked by reading, not compiled, not run on hardware
//
#include <hip/hip_runtime.h>
#include <hip/hip_bf16.h>

typedef __attribute__((ext_vector_type(16))) _Float16 v16h;
typedef __attribute__((ext_vector_type(8)))  float    v8f;

#define IN_C      128
#define OUTF      64     // HEADS * OC
#define HEADS     8
#define OC        8
#define EDIM      4
#define NEG_SLOPE 0.2f
#define BN_EPS    1e-5f

// ---------------------------------------------------------------------------
// utility: zero a float buffer
// ---------------------------------------------------------------------------
__global__ void zero_f32(float* __restrict__ p, int n) {
  int i = blockIdx.x * blockDim.x + threadIdx.x;
  if (i < n) p[i] = 0.0f;
}

// ---------------------------------------------------------------------------
// monotone float <-> uint key for atomic segment-max
// ---------------------------------------------------------------------------
__device__ __forceinline__ unsigned fkey(float f) {
  unsigned u = __float_as_uint(f);
  return (u & 0x80000000u) ? ~u : (u | 0x80000000u);
}
__device__ __forceinline__ float funkey(unsigned k) {
  unsigned b = (k & 0x80000000u) ? (k & 0x7fffffffu) : ~k;
  return __uint_as_float(b);
}

// ---------------------------------------------------------------------------
// K1: h = x @ W  and  ident = x @ res_W + res_b   (WMMA f16 -> f32)
// Each wave computes a 16x64 tile of both outputs.
// W/res_W are staged in LDS PRE-SWIZZLED into the WMMA B-fragment layout so
// each lane's 16 halves of a 32x16 B fragment are one contiguous 32B v16h
// (-> two ds_load_b128, no scalar half loads, no packing movs).
//   fragment f = (k/32)*4 + (n/16); element (lane l, half j) holds
//   B[k = (l>>4)*16 + j, n = l&15]  ->  LDS offset ((f*32 + l)*16 + j)
// ---------------------------------------------------------------------------
__global__ __launch_bounds__(128) void node_gemm_kernel(
    const float* __restrict__ x, const float* __restrict__ W,
    const float* __restrict__ resW, const float* __restrict__ resb,
    float* __restrict__ h, float* __restrict__ ident, int n_tiles) {
  __shared__ _Float16 sW[IN_C * OUTF];   // 16 KB, swizzled fragments
  __shared__ _Float16 sR[IN_C * OUTF];   // 16 KB, swizzled fragments

  // coalesced global read (row-major), scattered swizzle write to LDS
  for (int s = threadIdx.x; s < IN_C * OUTF; s += blockDim.x) {
    const int k    = s >> 6;                       // 0..127
    const int n    = s & 63;                       // 0..63
    const int kc_i = k >> 5;                       // k-chunk 0..3
    const int krem = k & 31;
    const int lane = ((krem >> 4) << 4) | (n & 15);
    const int j    = krem & 15;
    const int f    = kc_i * 4 + (n >> 4);
    const int dst  = ((f << 5) | lane) * 16 + j;
    sW[dst] = (_Float16)W[s];
    sR[dst] = (_Float16)resW[s];
  }
  __syncthreads();

  const int wave  = threadIdx.x >> 5;    // 0..3
  const int lane  = threadIdx.x & 31;
  const int r     = lane & 15;           // A row within tile
  const int khalf = lane >> 4;           // which K half this lane holds

  for (int tile = blockIdx.x * 4 + wave; tile < n_tiles; tile += gridDim.x * 4) {
    const int m0 = tile * 16;
    v8f accW[4], accR[4];
    const v8f vzero = {0.f, 0.f, 0.f, 0.f, 0.f, 0.f, 0.f, 0.f};
#pragma unroll
    for (int t = 0; t < 4; t++) { accW[t] = vzero; accR[t] = vzero; }

#pragma unroll
    for (int kc_i = 0; kc_i < 4; kc_i++) {
      const int kc = kc_i * 32;
      // A fragment (16x32 f16): lane holds K = khalf*16..+15
      const float4* xr =
          (const float4*)(x + (size_t)(m0 + r) * IN_C + kc + khalf * 16);
      const float4 p0 = xr[0], p1 = xr[1];   // K + 0..7
      const float4 q0 = xr[2], q1 = xr[3];   // K + 8..15
      v16h a;
      a[0]  = (_Float16)p0.x; a[1]  = (_Float16)p0.y;
      a[2]  = (_Float16)p0.z; a[3]  = (_Float16)p0.w;
      a[4]  = (_Float16)p1.x; a[5]  = (_Float16)p1.y;
      a[6]  = (_Float16)p1.z; a[7]  = (_Float16)p1.w;
      a[8]  = (_Float16)q0.x; a[9]  = (_Float16)q0.y;
      a[10] = (_Float16)q0.z; a[11] = (_Float16)q0.w;
      a[12] = (_Float16)q1.x; a[13] = (_Float16)q1.y;
      a[14] = (_Float16)q1.z; a[15] = (_Float16)q1.w;

#pragma unroll
      for (int t = 0; t < 4; t++) {
        const int f = kc_i * 4 + t;
        const v16h bW = *(const v16h*)(sW + ((size_t)((f << 5) | lane) * 16));
        const v16h bR = *(const v16h*)(sR + ((size_t)((f << 5) | lane) * 16));
        accW[t] = __builtin_amdgcn_wmma_f32_16x16x32_f16(
            false, a, false, bW, (short)0, accW[t], false, false);
        accR[t] = __builtin_amdgcn_wmma_f32_16x16x32_f16(
            false, a, false, bR, (short)0, accR[t], false, false);
      }
    }

    // C/D layout: element v of lane l -> row v + 8*(l>>4), col l&15
    const int mbase = khalf * 8;
#pragma unroll
    for (int t = 0; t < 4; t++) {
      const int n0 = t * 16;
#pragma unroll
      for (int v = 0; v < 8; v++) {
        const int m = m0 + mbase + v;
        const int c = n0 + r;
        h[(size_t)m * OUTF + c]     = accW[t][v];
        ident[(size_t)m * OUTF + c] = accR[t][v] + resb[c];
      }
    }
  }
}

// ---------------------------------------------------------------------------
// K2: per-node attention scalars a_src, a_dst  (one thread per (node, head))
// ---------------------------------------------------------------------------
__global__ void att_kernel(const float* __restrict__ h,
                           const float* __restrict__ att_src,
                           const float* __restrict__ att_dst,
                           float* __restrict__ asrc, float* __restrict__ adst,
                           int n_nodes) {
  int i = blockIdx.x * blockDim.x + threadIdx.x;
  if (i >= n_nodes * HEADS) return;
  int node = i >> 3, hd = i & 7;
  const float* hp = h + (size_t)node * OUTF + hd * OC;
  float s = 0.f, d = 0.f;
#pragma unroll
  for (int c = 0; c < OC; c++) {
    float v = hp[c];
    s += v * att_src[hd * OC + c];
    d += v * att_dst[hd * OC + c];
  }
  asrc[i] = s;
  adst[i] = d;
}

// ---------------------------------------------------------------------------
// K_pre: collapse edge attention:  wE[k][h] = sum_c W_edge[k][h*8+c]*att_edge[h][c]
// ---------------------------------------------------------------------------
__global__ void wedge_kernel(const float* __restrict__ W_edge,
                             const float* __restrict__ att_edge,
                             float* __restrict__ wE) {
  int t = threadIdx.x;
  if (t >= EDIM * HEADS) return;
  int k = t >> 3, hd = t & 7;
  float s = 0.f;
#pragma unroll
  for (int c = 0; c < OC; c++)
    s += W_edge[k * (HEADS * OC) + hd * OC + c] * att_edge[hd * OC + c];
  wE[k * HEADS + hd] = s;
}

// ---------------------------------------------------------------------------
// edge logit (recomputed in each softmax pass; compute is free vs. bandwidth)
// ---------------------------------------------------------------------------
__device__ __forceinline__ float edge_logit(int e, int hd, int src, int dst,
                                            const float* __restrict__ ea,
                                            const float* __restrict__ wE,
                                            const float* __restrict__ asrc,
                                            const float* __restrict__ adst) {
  float aE = 0.f;
#pragma unroll
  for (int k = 0; k < EDIM; k++) aE += ea[(size_t)e * EDIM + k] * wE[k * HEADS + hd];
  float l = asrc[src * HEADS + hd] + adst[dst * HEADS + hd] + aE;
  return l > 0.f ? l : NEG_SLOPE * l;
}

// K3: segment max via monotone-key u32 atomicMax
__global__ void edge_max_kernel(const int* __restrict__ ei, const float* __restrict__ ea,
                                const float* __restrict__ wE,
                                const float* __restrict__ asrc,
                                const float* __restrict__ adst,
                                unsigned* __restrict__ mkey, int n_edges) {
  int i = blockIdx.x * blockDim.x + threadIdx.x;
  if (i >= n_edges * HEADS) return;
  int e = i >> 3, hd = i & 7;
  int src = ei[e], dst = ei[n_edges + e];
  float l = edge_logit(e, hd, src, dst, ea, wE, asrc, adst);
  atomicMax(&mkey[dst * HEADS + hd], fkey(l));
}

// K4: denom = segment_sum(exp(logit - m[dst]))
__global__ void edge_sum_kernel(const int* __restrict__ ei, const float* __restrict__ ea,
                                const float* __restrict__ wE,
                                const float* __restrict__ asrc,
                                const float* __restrict__ adst,
                                const unsigned* __restrict__ mkey,
                                float* __restrict__ denom, int n_edges) {
  int i = blockIdx.x * blockDim.x + threadIdx.x;
  if (i >= n_edges * HEADS) return;
  int e = i >> 3, hd = i & 7;
  int src = ei[e], dst = ei[n_edges + e];
  float l = edge_logit(e, hd, src, dst, ea, wE, asrc, adst);
  float mraw = funkey(mkey[dst * HEADS + hd]);
  float m = __builtin_isfinite(mraw) ? mraw : 0.f;
  float num = __expf(l - m);
  atomicAdd(&denom[dst * HEADS + hd], num);
}

// K5: out[dst] += alpha * h[src]
__global__ void edge_scatter_kernel(const int* __restrict__ ei, const float* __restrict__ ea,
                                    const float* __restrict__ wE,
                                    const float* __restrict__ asrc,
                                    const float* __restrict__ adst,
                                    const unsigned* __restrict__ mkey,
                                    const float* __restrict__ denom,
                                    const float* __restrict__ h,
                                    float* __restrict__ out, int n_edges) {
  int i = blockIdx.x * blockDim.x + threadIdx.x;
  if (i >= n_edges * HEADS) return;
  int e = i >> 3, hd = i & 7;
  int src = ei[e], dst = ei[n_edges + e];
  float l = edge_logit(e, hd, src, dst, ea, wE, asrc, adst);
  float mraw = funkey(mkey[dst * HEADS + hd]);
  float m = __builtin_isfinite(mraw) ? mraw : 0.f;
  float num = __expf(l - m);
  float alpha = num / (denom[dst * HEADS + hd] + 1e-16f);
  const float* hs = h + (size_t)src * OUTF + hd * OC;
  float* op = out + (size_t)dst * OUTF + hd * OC;
#pragma unroll
  for (int c = 0; c < OC; c++) atomicAdd(&op[c], alpha * hs[c]);
}

// ---------------------------------------------------------------------------
// K6: BN batch statistics (sum, sumsq per channel) via LDS + global atomics
// ---------------------------------------------------------------------------
__global__ void bn_stats_kernel(const float* __restrict__ gat,
                                const float* __restrict__ bias,
                                float* __restrict__ stats, int total) {
  __shared__ float ssum[OUTF], ssq[OUTF];
  if (threadIdx.x < OUTF) { ssum[threadIdx.x] = 0.f; ssq[threadIdx.x] = 0.f; }
  __syncthreads();
  for (int i = blockIdx.x * blockDim.x + threadIdx.x; i < total;
       i += gridDim.x * blockDim.x) {
    int c = i & (OUTF - 1);
    float t = gat[i] + bias[c];
    atomicAdd(&ssum[c], t);
    atomicAdd(&ssq[c], t * t);
  }
  __syncthreads();
  if (threadIdx.x < OUTF) {
    atomicAdd(&stats[threadIdx.x], ssum[threadIdx.x]);
    atomicAdd(&stats[OUTF + threadIdx.x], ssq[threadIdx.x]);
  }
}

// ---------------------------------------------------------------------------
// K7: BN normalize + ELU + residual, in-place on d_out
// ---------------------------------------------------------------------------
__global__ void final_kernel(float* __restrict__ gat_out,
                             const float* __restrict__ ident,
                             const float* __restrict__ bias,
                             const float* __restrict__ stats,
                             const float* __restrict__ bn_w,
                             const float* __restrict__ bn_b,
                             int n_nodes, int total) {
  int i = blockIdx.x * blockDim.x + threadIdx.x;
  if (i >= total) return;
  int c = i & (OUTF - 1);
  float invN = 1.0f / (float)n_nodes;
  float mean = stats[c] * invN;
  float var  = stats[OUTF + c] * invN - mean * mean;
  float t = gat_out[i] + bias[c];
  float y = (t - mean) * rsqrtf(var + BN_EPS) * bn_w[c] + bn_b[c];
  y = y > 0.f ? y : (__expf(y) - 1.0f);
  gat_out[i] = y + ident[i];
}

// ---------------------------------------------------------------------------
extern "C" void kernel_launch(void* const* d_in, const int* in_sizes, int n_in,
                              void* d_out, int out_size, void* d_ws, size_t ws_size,
                              hipStream_t stream) {
  const float* x        = (const float*)d_in[0];
  const int*   ei       = (const int*)  d_in[1];
  const float* ea       = (const float*)d_in[2];
  const float* W        = (const float*)d_in[3];
  const float* att_src  = (const float*)d_in[4];
  const float* att_dst  = (const float*)d_in[5];
  const float* W_edge   = (const float*)d_in[6];
  const float* att_edge = (const float*)d_in[7];
  const float* bias     = (const float*)d_in[8];
  const float* bn_w     = (const float*)d_in[9];
  const float* bn_b     = (const float*)d_in[10];
  const float* res_W    = (const float*)d_in[11];
  const float* res_b    = (const float*)d_in[12];
  float* out = (float*)d_out;

  const int n_nodes = in_sizes[0] / IN_C;
  const int n_edges = in_sizes[1] / 2;
  const int total   = n_nodes * OUTF;
  const int nh      = n_nodes * HEADS;
  const int eh      = n_edges * HEADS;

  // workspace layout (floats)
  float*    ws    = (float*)d_ws;
  float*    h     = ws;                              // N*64
  float*    ident = h     + (size_t)total;           // N*64
  float*    asrc  = ident + (size_t)total;           // N*8
  float*    adst  = asrc  + (size_t)nh;              // N*8
  unsigned* mkey  = (unsigned*)(adst + (size_t)nh);  // N*8
  float*    denom = (float*)mkey + (size_t)nh;       // N*8
  float*    stats = denom + (size_t)nh;              // 128
  float*    wE    = stats + 2 * OUTF;                // 32

  // zero accumulators (harness poisons buffers; must re-zero every call)
  zero_f32<<<(total + 255) / 256, 256, 0, stream>>>(out, total);
  zero_f32<<<(nh + 255) / 256, 256, 0, stream>>>((float*)mkey, nh); // key 0 == -inf-ish
  zero_f32<<<(nh + 255) / 256, 256, 0, stream>>>(denom, nh);
  zero_f32<<<1, 256, 0, stream>>>(stats, 2 * OUTF);

  const int n_tiles = (n_nodes + 15) / 16;
  node_gemm_kernel<<<512, 128, 0, stream>>>(x, W, res_W, res_b, h, ident, n_tiles);
  wedge_kernel<<<1, 32, 0, stream>>>(W_edge, att_edge, wE);
  att_kernel<<<(nh + 255) / 256, 256, 0, stream>>>(h, att_src, att_dst, asrc, adst, n_nodes);

  edge_max_kernel<<<(eh + 255) / 256, 256, 0, stream>>>(ei, ea, wE, asrc, adst, mkey, n_edges);
  edge_sum_kernel<<<(eh + 255) / 256, 256, 0, stream>>>(ei, ea, wE, asrc, adst, mkey, denom, n_edges);
  edge_scatter_kernel<<<(eh + 255) / 256, 256, 0, stream>>>(ei, ea, wE, asrc, adst, mkey,
                                                            denom, h, out, n_edges);

  bn_stats_kernel<<<1024, 256, 0, stream>>>(out, bias, stats, total);
  final_kernel<<<(total + 255) / 256, 256, 0, stream>>>(out, ident, bias, stats,
                                                        bn_w, bn_b, n_nodes, total);
}
